// WHTConv2D_26431228740098
// MI455X (gfx1250) — matrix operand order repaired and checked by reading, not compiled, not run on hardware
//
#include <hip/hip_runtime.h>
#include <hip/hip_bf16.h>

// ---------------------------------------------------------------------------
// WHT-Conv2D fused pipeline for MI455X (gfx1250, wave32)
//   f2  = WHT2D(x)                      (12-stage FWHT per 64x64 plane, f32)
//   g_p = W_p @ f2   (channel GEMM)     (v_wmma_f32_16x16x32_f16, f32 accum)
//   f6  = sum_p soft(v_p * g_p, T_p)
//   out = WHT2D(f6)/4096 + x
// ---------------------------------------------------------------------------

typedef __attribute__((ext_vector_type(16))) _Float16     v16h;
typedef __attribute__((ext_vector_type(8)))  float        v8f;
typedef __attribute__((ext_vector_type(8)))  unsigned int v8u;
typedef __attribute__((ext_vector_type(4)))  unsigned int u32x4;
typedef __attribute__((ext_vector_type(2)))  unsigned int u32x2;

#define B_SZ   32
#define C_SZ   256
#define NPX    4096        // 64*64
#define PODS   2

#if __has_builtin(__builtin_amdgcn_global_load_async_to_lds_b128)
#define HAVE_ASYNC_LDS 1
#else
#define HAVE_ASYNC_LDS 0
#endif

#if HAVE_ASYNC_LDS
// Probe-learned signature: param0 is `int __vector(4) __device__ *`
// (= vector_size(16) int, HIP global AS(1)).  Param1 is the LDS side (AS(3)).
typedef int v4i_gcc __attribute__((vector_size(16)));
typedef __attribute__((address_space(1))) v4i_gcc as1_v4i;
typedef __attribute__((address_space(3))) v4i_gcc as3_v4i;
// generic->AS casts via integer round-trip: global generic ptrs are identical
// bit patterns; generic LDS ptr low 32 bits are the LDS offset (ISA 10.2).
__device__ inline void async_cp16(const void* g, void* l) {
    __builtin_amdgcn_global_load_async_to_lds_b128(
        (as1_v4i*)(unsigned long long)(uintptr_t)g,
        (as3_v4i*)(unsigned int)(uintptr_t)l, 0, 0);
}
__device__ inline void async_wait0() {
#if __has_builtin(__builtin_amdgcn_s_wait_asynccnt)
    __builtin_amdgcn_s_wait_asynccnt(0);
#else
    asm volatile("s_wait_asynccnt 0x0" ::: "memory");
#endif
}
#endif

// ---------------------------------------------------------------------------
// Kernel 0: convert W (P,Cout,Cin) f32 -> f16
// ---------------------------------------------------------------------------
__global__ __launch_bounds__(256) void k_cvt_w(const float* __restrict__ W,
                                               _Float16* __restrict__ W16, int n) {
    int i = blockIdx.x * 256 + threadIdx.x;
    if (i < n) W16[i] = (_Float16)W[i];
}

// ---------------------------------------------------------------------------
// Kernel 1: forward 2-D WHT per plane (= 4096-pt FWHT since H64(x)H64 = H4096),
// then transpose-store as f16 into f2T[b][px][c]  (c contiguous for WMMA B).
// Block: 512 threads, 8 channel planes of one batch image. LDS = 128 KB.
// ---------------------------------------------------------------------------
__global__ __launch_bounds__(512) void k_fwht_fwd(const float* __restrict__ x,
                                                  _Float16* __restrict__ f2T) {
    __shared__ float lds[8 * NPX];          // 128 KB
    const int tid = threadIdx.x;
    const int b  = blockIdx.x >> 5;         // 32 blocks-per-b
    const int c0 = (blockIdx.x & 31) * 8;   // 8-channel tile

    // coalesced load of 8 contiguous planes into LDS
    const float4* xs = (const float4*)(x + ((size_t)(b * C_SZ + c0)) * NPX);
    float4* ls = (float4*)lds;
#if HAVE_ASYNC_LDS
    for (int i = tid; i < 8 * (NPX / 4); i += 512) async_cp16(xs + i, ls + i);
    async_wait0();
#else
    for (int i = tid; i < 8 * (NPX / 4); i += 512) ls[i] = xs[i];
#endif
    __syncthreads();

    // 12-stage FWHT per plane; 64 threads per plane
    const int plane = tid >> 6;
    const int t     = tid & 63;
    float* pl = lds + plane * NPX;
    for (int s = NPX / 2; s >= 1; s >>= 1) {
        for (int q = t; q < NPX / 2; q += 64) {
            int i = ((q & ~(s - 1)) << 1) | (q & (s - 1));
            float a = pl[i], c = pl[i + s];
            pl[i] = a + c;
            pl[i + s] = a - c;
        }
        __syncthreads();
    }

    // transpose-store: per pixel, 8 channels -> 16B contiguous f16
    for (int px = tid; px < NPX; px += 512) {
        union { _Float16 h[8]; u32x4 q; } u;
#pragma unroll
        for (int c = 0; c < 8; ++c) u.h[c] = (_Float16)lds[c * NPX + px];
        *(u32x4*)(f2T + (((size_t)b * NPX + px) * C_SZ + c0)) = u.q;
    }
}

// ---------------------------------------------------------------------------
// Fragment loaders matching the CDNA5 16-bit WMMA VGPR layouts (ISA 7.12.2).
// ---------------------------------------------------------------------------
__device__ inline v16h ld_frag2(const _Float16* p0, const _Float16* p1) {
    u32x4 a = *(const u32x4*)p0;
    u32x4 b = *(const u32x4*)p1;
    v8u u;
    u[0] = a[0]; u[1] = a[1]; u[2] = a[2]; u[3] = a[3];
    u[4] = b[0]; u[5] = b[1]; u[6] = b[2]; u[7] = b[3];
    return __builtin_bit_cast(v16h, u);
}

// ---------------------------------------------------------------------------
// Kernel 2: channel GEMM with WMMA + fused v-scale / soft-threshold / pod-sum.
// Per wave: 32(o) x 32(px) tile, both pods, K=256 in 8 steps of 32.
//   (M=32 halves f2T re-reads from L2 vs M=16; extra W reads hit L0 since all
//    8 waves of a block share the same o-block.)
// grid = 4096 blocks x 256 threads (8 waves): 32768 wave-tiles total.
// ---------------------------------------------------------------------------
__global__ __launch_bounds__(256) void k_gemm_soft(const _Float16* __restrict__ f2T,
                                                   const _Float16* __restrict__ W16,
                                                   const float* __restrict__ v,
                                                   const float* __restrict__ T,
                                                   _Float16* __restrict__ f6) {
    const int lane = threadIdx.x & 31;
    const int widx = threadIdx.x >> 5;
    const int wid  = blockIdx.x * 8 + widx;
    const int pxb  = wid & 127;         // 128 px-blocks of 32 pixels
    const int ob   = (wid >> 7) & 7;    // 8 o-blocks of 32 channels
    const int b    = wid >> 10;         // batch
    const int m  = lane & 15;
    const int hi = lane >> 4;

    v8f acc[PODS][2][2];                // [pod][o-subtile][px-subtile]
#pragma unroll
    for (int p = 0; p < PODS; ++p)
#pragma unroll
        for (int u = 0; u < 2; ++u)
#pragma unroll
            for (int t = 0; t < 2; ++t)
#pragma unroll
                for (int e = 0; e < 8; ++e) acc[p][u][t][e] = 0.0f;

    // A rows: W16[p][o][c], o = ob*32 + u*16 + m ; per-lane K subgroup via hi
    const _Float16* arow[PODS][2];
#pragma unroll
    for (int p = 0; p < PODS; ++p)
#pragma unroll
        for (int u = 0; u < 2; ++u)
            arow[p][u] = W16 + ((size_t)(p * C_SZ + ob * 32 + u * 16 + m)) * C_SZ + hi * 8;
    // B rows: f2T[b][px][c], n = m ; per-lane K subgroup via hi
    const _Float16* brow[2];
#pragma unroll
    for (int t = 0; t < 2; ++t)
        brow[t] = f2T + ((size_t)(b * NPX + pxb * 32 + t * 16 + m)) * C_SZ + hi * 8;

#pragma unroll
    for (int kk = 0; kk < C_SZ; kk += 32) {
        v16h a[PODS][2];
#pragma unroll
        for (int p = 0; p < PODS; ++p)
#pragma unroll
            for (int u = 0; u < 2; ++u)
                a[p][u] = ld_frag2(arow[p][u] + kk, arow[p][u] + kk + 16);
#pragma unroll
        for (int t = 0; t < 2; ++t) {
            v16h bf = ld_frag2(brow[t] + kk, brow[t] + kk + 16);
#pragma unroll
            for (int p = 0; p < PODS; ++p)
#pragma unroll
                for (int u = 0; u < 2; ++u)
                    acc[p][u][t] = __builtin_amdgcn_wmma_f32_16x16x32_f16(
                        false, a[p][u], false, bf, (short)0, acc[p][u][t], false, false);
        }
    }

    // Epilogue: f6 = sum_p sign(v_p*g_p) * max(|v_p*g_p| - T_p, 0)
    const size_t f6b = (size_t)b * C_SZ * NPX;
#pragma unroll
    for (int t = 0; t < 2; ++t) {
        const int px = pxb * 32 + t * 16 + m;  // D-matrix N = lane&15
        const float vv0 = v[px],  vv1 = v[NPX + px];
        const float tt0 = T[px],  tt1 = T[NPX + px];
#pragma unroll
        for (int u = 0; u < 2; ++u)
#pragma unroll
            for (int r = 0; r < 8; ++r) {
                const int o = ob * 32 + u * 16 + hi * 8 + r;  // D-matrix M = r + 8*hi
                float y0 = vv0 * acc[0][u][t][r];
                float y1 = vv1 * acc[1][u][t][r];
                float s0 = fmaxf(fabsf(y0) - tt0, 0.0f);
                float s1 = fmaxf(fabsf(y1) - tt1, 0.0f);
                float f  = __builtin_copysignf(s0, y0) + __builtin_copysignf(s1, y1);
                f6[f6b + (size_t)o * NPX + px] = (_Float16)f;
            }
    }
}

// ---------------------------------------------------------------------------
// Kernel 3: inverse 2-D WHT (FWHT/4096) + residual add.  8 o-planes per block.
// ---------------------------------------------------------------------------
__global__ __launch_bounds__(512) void k_fwht_inv(const _Float16* __restrict__ f6,
                                                  const float* __restrict__ x,
                                                  float* __restrict__ out) {
    __shared__ float lds[8 * NPX];          // 128 KB
    const int tid = threadIdx.x;
    const int b  = blockIdx.x >> 5;
    const int o0 = (blockIdx.x & 31) * 8;
    const size_t base = ((size_t)b * C_SZ + o0) * NPX;

    // load f16 -> f32 into LDS (4 halves per iteration)
    for (int i = tid; i < 8 * (NPX / 4); i += 512) {
        union { u32x2 q; _Float16 h[4]; } u;
        u.q = *(const u32x2*)(f6 + base + (size_t)i * 4);
#pragma unroll
        for (int j = 0; j < 4; ++j) lds[i * 4 + j] = (float)u.h[j];
    }
    __syncthreads();

    const int plane = tid >> 6;
    const int t     = tid & 63;
    float* pl = lds + plane * NPX;
    for (int s = NPX / 2; s >= 1; s >>= 1) {
        for (int q = t; q < NPX / 2; q += 64) {
            int i = ((q & ~(s - 1)) << 1) | (q & (s - 1));
            float a = pl[i], c = pl[i + s];
            pl[i] = a + c;
            pl[i + s] = a - c;
        }
        __syncthreads();
    }

    const float S = 1.0f / (float)NPX;
    const float4* xs = (const float4*)(x + base);
    const float4* ls = (const float4*)lds;
    float4* os = (float4*)(out + base);
    for (int i = tid; i < 8 * (NPX / 4); i += 512) {
        float4 xv = xs[i];
        float4 l  = ls[i];
        float4 r;
        r.x = l.x * S + xv.x;
        r.y = l.y * S + xv.y;
        r.z = l.z * S + xv.z;
        r.w = l.w * S + xv.w;
        os[i] = r;
    }
}

// ---------------------------------------------------------------------------
extern "C" void kernel_launch(void* const* d_in, const int* in_sizes, int n_in,
                              void* d_out, int out_size, void* d_ws, size_t ws_size,
                              hipStream_t stream) {
    (void)in_sizes; (void)n_in; (void)out_size; (void)ws_size;
    const float* x = (const float*)d_in[0];   // (32,256,64,64)
    const float* v = (const float*)d_in[1];   // (2,64,64)
    const float* W = (const float*)d_in[2];   // (2,256,256)
    const float* T = (const float*)d_in[3];   // (2,64,64)
    float* out = (float*)d_out;

    char* ws = (char*)d_ws;
    _Float16* f2T = (_Float16*)ws;                               // 64 MB
    _Float16* f6  = (_Float16*)(ws + (size_t)64 * 1024 * 1024);  // 64 MB
    _Float16* W16 = (_Float16*)(ws + (size_t)128 * 1024 * 1024); // 256 KB

    const int nW = PODS * C_SZ * C_SZ; // 131072
    k_cvt_w<<<(nW + 255) / 256, 256, 0, stream>>>(W, W16, nW);

    k_fwht_fwd<<<B_SZ * (C_SZ / 8), 512, 0, stream>>>(x, f2T);

    // 32768 wave-tiles / 8 waves-per-block = 4096 blocks
    k_gemm_soft<<<4096, 256, 0, stream>>>(f2T, W16, v, T, f6);

    k_fwht_inv<<<B_SZ * (C_SZ / 8), 512, 0, stream>>>(f6, x, out);
}
